// GNNEncoder_3350074491177
// MI455X (gfx1250) — matrix operand order repaired and checked by reading, not compiled, not run on hardware
//
#include <hip/hip_runtime.h>
#include <hip/hip_bf16.h>
#include <math.h>

// ---------------- problem constants ----------------
#define NN      50000        // N_NODES
#define NE      600000       // N_EDGES
#define HD      128          // H
#define NET     3            // N_ETYPES
#define NG      64           // N_GRAPHS
#define TH3     384          // 3*H
#define NLAYERS 2
#define NSTEPS  2

typedef __attribute__((ext_vector_type(8)))  float   v8f;
typedef __attribute__((ext_vector_type(16))) __bf16  v16bf;
typedef __attribute__((ext_vector_type(8)))  __bf16  v8bf;
typedef __attribute__((ext_vector_type(4)))  __bf16  v4bf;

// ---------------- WMMA helpers (CDNA5 16x16x32 bf16, fp32 accum) ----------------
__device__ __forceinline__ v8f wmma_bf16(v16bf a, v16bf b, v8f c) {
  // (neg_a, A, neg_b, B, c_mod, C, reuse_a, reuse_b)
  return __builtin_amdgcn_wmma_f32_16x16x32_bf16(false, a, false, b, (short)0, c, false, false);
}

// A-matrix 16x32 bf16 fragment (matches B-fragment K order):
// lanes 0-15: M=lane, elems 0-15 -> K 0..15
// lanes16-31: M=lane-16, elems 0-15 -> K 16..31
__device__ __forceinline__ v16bf load_a_frag(const __bf16* rowptr, int kc, int lane) {
  const int koff = (lane < 16) ? 0 : 16;
  return *(const v16bf*)(rowptr + 32 * kc + koff);
}

// B-matrix 32x16 bf16 fragment: col = lane&15; lanes 0-15 hold K 0..15,
// lanes 16-31 hold K 16..31. colptr points at column-major weight row (16 contiguous K).
__device__ __forceinline__ v16bf load_b_frag(const __bf16* colptr, int kc, int lane) {
  const int koff = (lane < 16) ? 0 : 16;
  return *(const v16bf*)(colptr + 32 * kc + koff);
}

__device__ __forceinline__ float sigmoidf_(float x) { return 1.0f / (1.0f + __expf(-x)); }

// ---------------- small utility kernels ----------------
__global__ void zero_kernel(float* __restrict__ p, int n) {
  int i = blockIdx.x * blockDim.x + threadIdx.x;
  if (i < n) p[i] = 0.0f;
}

__global__ void cvt_f2b_kernel(const float* __restrict__ x, __bf16* __restrict__ y, int n) {
  int i = blockIdx.x * blockDim.x + threadIdx.x;
  if (i < n) y[i] = (__bf16)x[i];
}

// Convert/transposed-pack all weights to bf16. 2*3*128*128 == 2*384*128 == 98304.
__global__ void cvt_weights_kernel(const float* __restrict__ Wl,
                                   const float* __restrict__ Wih,
                                   const float* __restrict__ Whh,
                                   __bf16* __restrict__ WlT,
                                   __bf16* __restrict__ Wihb,
                                   __bf16* __restrict__ Whhb) {
  int i = blockIdx.x * blockDim.x + threadIdx.x;
  const int NW = NLAYERS * NET * HD * HD;   // 98304
  if (i >= NW) return;
  int h  = i & (HD - 1);
  int o  = (i >> 7) & (HD - 1);
  int le = i >> 14;
  // WlT[le][o][h] = Wl[le][h][o]  (column-major for B-fragment loads)
  WlT[i]  = (__bf16)Wl[((size_t)le * HD + h) * HD + o];
  // W_ih / W_hh are used as feat @ W.T -> natural [j][h] layout is already col-major
  Wihb[i] = (__bf16)Wih[i];
  Whhb[i] = (__bf16)Whh[i];
}

// h = emb[text_idx]; also write bf16 copy. One wave per node, 4 cols/lane.
__global__ void embed_kernel(const int* __restrict__ idx, const float* __restrict__ emb,
                             float* __restrict__ h, __bf16* __restrict__ hb) {
  int tid = blockIdx.x * blockDim.x + threadIdx.x;
  int n = tid >> 5, lane = tid & 31;
  if (n >= NN) return;
  const float* e = emb + (size_t)idx[n] * HD + lane * 4;
  size_t o = (size_t)n * HD + lane * 4;
#pragma unroll
  for (int k = 0; k < 4; ++k) { float v = e[k]; h[o + k] = v; hb[o + k] = (__bf16)v; }
}

// ---------------- etype linear: Whb[e][n][:] = bf16( feat @ W[e] + b[e] ) ----------------
__global__ __launch_bounds__(32)
void etype_gemm_kernel(const __bf16* __restrict__ hb,
                       const __bf16* __restrict__ WlT,   // [3][128][128] col-major (this layer)
                       const float*  __restrict__ bl,    // [3][128]      (this layer)
                       __bf16* __restrict__ Whb) {       // [3][NN][128]
  const int lane = threadIdx.x;
  const int n0 = blockIdx.x * 16;
  const __bf16* arow = hb + (size_t)(n0 + (lane & 15)) * HD;
  v16bf A[4];
#pragma unroll
  for (int kc = 0; kc < 4; ++kc) A[kc] = load_a_frag(arow, kc, lane);
  const int col = lane & 15;
  const int rbase = (lane < 16) ? 0 : 8;
  for (int e = 0; e < NET; ++e) {
    for (int c = 0; c < 8; ++c) {
      const __bf16* bcol = WlT + ((size_t)e * HD + c * 16 + col) * HD;
      v8f acc = {};
#pragma unroll
      for (int kc = 0; kc < 4; ++kc)
        acc = wmma_bf16(A[kc], load_b_frag(bcol, kc, lane), acc);
      const float bias = bl[e * HD + c * 16 + col];
      __bf16* out = Whb + ((size_t)e * NN + n0) * HD + c * 16 + col;
#pragma unroll
      for (int r = 0; r < 8; ++r)
        out[(size_t)(rbase + r) * HD] = (__bf16)(acc[r] + bias);
    }
  }
}

// ---------------- per-edge gather + scatter-add: a[dst] += Wh[etype][src] ----------------
__global__ __launch_bounds__(256)
void edge_scatter_kernel(const int* __restrict__ src, const int* __restrict__ dst,
                         const int* __restrict__ et, const __bf16* __restrict__ Whb,
                         float* __restrict__ a) {
  int tid = blockIdx.x * 256 + threadIdx.x;
  int e = tid >> 5, lane = tid & 31;
  if (e >= NE) return;
  int s = src[e], d = dst[e], ty = et[e];
  const __bf16* m = Whb + ((size_t)ty * NN + s) * HD + lane * 4;
  float* out = a + (size_t)d * HD + lane * 4;
  v4bf mv = *(const v4bf*)m;
#pragma unroll
  for (int k = 0; k < 4; ++k) atomicAdd(&out[k], (float)mv[k]);
}

// ---------------- fused GRU: both GEMMs (a@Wih.T, h@Whh.T) + gates, in-place h ----------------
__global__ __launch_bounds__(32)
void gru_fused_kernel(const __bf16* __restrict__ ab,      // [NN][128] bf16 aggregated msgs
                      const __bf16* __restrict__ Wihb,    // [384][128] col-major (this layer)
                      const __bf16* __restrict__ Whhb,    // [384][128] col-major (this layer)
                      const float*  __restrict__ bih,     // [384]
                      const float*  __restrict__ bhh,     // [384]
                      float*  __restrict__ h,             // [NN][128] in/out
                      __bf16* __restrict__ hb) {          // [NN][128] in/out
  const int lane = threadIdx.x;
  const int n0 = blockIdx.x * 16;
  const __bf16* arow = ab + (size_t)(n0 + (lane & 15)) * HD;
  const __bf16* frow = hb + (size_t)(n0 + (lane & 15)) * HD;
  v16bf A[4], F[4];
#pragma unroll
  for (int kc = 0; kc < 4; ++kc) { A[kc] = load_a_frag(arow, kc, lane); F[kc] = load_a_frag(frow, kc, lane); }
  const int col = lane & 15;
  const int rbase = (lane < 16) ? 0 : 8;
  for (int t = 0; t < 8; ++t) {
    const int jr = t * 16 + col;
    const __bf16* Br = Wihb + (size_t)jr * HD;
    const __bf16* Bz = Wihb + (size_t)(HD + jr) * HD;
    const __bf16* Bn = Wihb + (size_t)(2 * HD + jr) * HD;
    const __bf16* Cr = Whhb + (size_t)jr * HD;
    const __bf16* Cz = Whhb + (size_t)(HD + jr) * HD;
    const __bf16* Cn = Whhb + (size_t)(2 * HD + jr) * HD;
    v8f ir = {}, iz = {}, inn = {}, hr = {}, hz = {}, hn = {};
#pragma unroll
    for (int kc = 0; kc < 4; ++kc) {
      ir  = wmma_bf16(A[kc], load_b_frag(Br, kc, lane), ir);
      iz  = wmma_bf16(A[kc], load_b_frag(Bz, kc, lane), iz);
      inn = wmma_bf16(A[kc], load_b_frag(Bn, kc, lane), inn);
      hr  = wmma_bf16(F[kc], load_b_frag(Cr, kc, lane), hr);
      hz  = wmma_bf16(F[kc], load_b_frag(Cz, kc, lane), hz);
      hn  = wmma_bf16(F[kc], load_b_frag(Cn, kc, lane), hn);
    }
    const float bir = bih[jr], biz = bih[HD + jr], bin = bih[2 * HD + jr];
    const float bhr = bhh[jr], bhz = bhh[HD + jr], bhn = bhh[2 * HD + jr];
#pragma unroll
    for (int r = 0; r < 8; ++r) {
      size_t o = (size_t)(n0 + rbase + r) * HD + t * 16 + col;
      float rg = sigmoidf_(ir[r] + bir + hr[r] + bhr);
      float zg = sigmoidf_(iz[r] + biz + hz[r] + bhz);
      float nc = tanhf(inn[r] + bin + rg * (hn[r] + bhn));
      float hnew = (1.0f - zg) * nc + zg * h[o];
      h[o] = hnew;
      hb[o] = (__bf16)hnew;
    }
  }
}

// ---------------- segment mean over graphs ----------------
__global__ void mean_accum_kernel(const float* __restrict__ h, const int* __restrict__ gid,
                                  float* __restrict__ sums, float* __restrict__ cnt) {
  int tid = blockIdx.x * blockDim.x + threadIdx.x;
  int n = tid >> 5, lane = tid & 31;
  if (n >= NN) return;
  int g = gid[n];
  int c = lane * 4;
#pragma unroll
  for (int k = 0; k < 4; ++k) atomicAdd(&sums[g * HD + c + k], h[(size_t)n * HD + c + k]);
  if (lane == 0) atomicAdd(&cnt[g], 1.0f);
}

__global__ void mean_div_kernel(const float* __restrict__ sums, const float* __restrict__ cnt,
                                float* __restrict__ agg, int coloff) {
  int i = blockIdx.x * blockDim.x + threadIdx.x;
  if (i >= NG * HD) return;
  int g = i >> 7, c = i & (HD - 1);
  float cc = cnt[g]; cc = cc < 1.0f ? 1.0f : cc;
  agg[(size_t)g * TH3 + coloff + c] = sums[i] / cc;
}

// ---------------- MLP head: res = relu(agg@W1+b1)@W2 + b2 ----------------
__global__ __launch_bounds__(128)
void head_kernel(const float* __restrict__ agg, const float* __restrict__ W1,
                 const float* __restrict__ b1, const float* __restrict__ W2,
                 const float* __restrict__ b2, float* __restrict__ out) {
  __shared__ float xs[HD];
  int g = blockIdx.x, j = threadIdx.x;
  float acc = b1[j];
  for (int k = 0; k < TH3; ++k) acc += agg[(size_t)g * TH3 + k] * W1[(size_t)k * HD + j];
  float x = acc > 0.0f ? acc : 0.0f;
  xs[j] = x * W2[j];
  __syncthreads();
  for (int s = 64; s > 0; s >>= 1) { if (j < s) xs[j] += xs[j + s]; __syncthreads(); }
  if (j == 0) out[g] = xs[0] + b2[0];
}

__global__ void copy_agg_kernel(const float* __restrict__ agg, float* __restrict__ out) {
  int i = blockIdx.x * blockDim.x + threadIdx.x;
  if (i < NG * TH3) out[i] = agg[i];
}

// ---------------- host launch ----------------
extern "C" void kernel_launch(void* const* d_in, const int* in_sizes, int n_in,
                              void* d_out, int out_size, void* d_ws, size_t ws_size,
                              hipStream_t stream) {
  const int*   text_idx = (const int*)d_in[0];
  const int*   src      = (const int*)d_in[1];
  const int*   dst      = (const int*)d_in[2];
  const int*   etype    = (const int*)d_in[3];
  const int*   gid      = (const int*)d_in[4];
  const float* emb      = (const float*)d_in[5];
  const float* Wl       = (const float*)d_in[6];
  const float* bl       = (const float*)d_in[7];
  const float* Wih      = (const float*)d_in[8];
  const float* Whh      = (const float*)d_in[9];
  const float* bih      = (const float*)d_in[10];
  const float* bhh      = (const float*)d_in[11];
  const float* W1       = (const float*)d_in[12];
  const float* b1       = (const float*)d_in[13];
  const float* W2       = (const float*)d_in[14];
  const float* b2       = (const float*)d_in[15];
  float* out = (float*)d_out;

  // carve workspace (256B aligned chunks), ~116 MB total
  char* wsp = (char*)d_ws;
  size_t off = 0;
  auto carve = [&](size_t bytes) -> void* {
    void* p = (void*)(wsp + off);
    off += (bytes + 255) & ~(size_t)255;
    return p;
  };
  __bf16* hb   = (__bf16*)carve((size_t)NN * HD * 2);
  __bf16* ab   = (__bf16*)carve((size_t)NN * HD * 2);
  __bf16* Whb  = (__bf16*)carve((size_t)NET * NN * HD * 2);
  float*  aagg = (float*) carve((size_t)NN * HD * 4);
  float*  hfp  = (float*) carve((size_t)NN * HD * 4);
  __bf16* WlTb = (__bf16*)carve((size_t)NLAYERS * NET * HD * HD * 2);
  __bf16* Wihb = (__bf16*)carve((size_t)NLAYERS * TH3 * HD * 2);
  __bf16* Whhb = (__bf16*)carve((size_t)NLAYERS * TH3 * HD * 2);
  float*  sums = (float*) carve((size_t)NG * HD * 4);
  float*  cnt  = (float*) carve((size_t)NG * 4);
  float*  agg  = (float*) carve((size_t)NG * TH3 * 4);

  const int NH = NN * HD;                 // 6,400,000
  const int GR_EL  = (NH + 255) / 256;    // element-wise grids
  const int GR_N32 = (NN * 32) / 256;     // 6250 (one wave per node)
  const int GR_E32 = (NE * 32) / 256;     // 75000 (one wave per edge)
  const int GR_TILE = NN / 16;            // 3125 node tiles

  // weights -> bf16 (transposed where needed)
  cvt_weights_kernel<<<(NLAYERS * NET * HD * HD + 255) / 256, 256, 0, stream>>>(
      Wl, Wih, Whh, WlTb, Wihb, Whhb);

  // h = emb[text_idx]
  embed_kernel<<<GR_N32, 256, 0, stream>>>(text_idx, emb, hfp, hb);

  // intermediate[0] = mean_nodes(h)
  zero_kernel<<<(NG * HD + 255) / 256, 256, 0, stream>>>(sums, NG * HD);
  zero_kernel<<<1, 256, 0, stream>>>(cnt, NG);
  mean_accum_kernel<<<GR_N32, 256, 0, stream>>>(hfp, gid, sums, cnt);
  mean_div_kernel<<<(NG * HD + 255) / 256, 256, 0, stream>>>(sums, cnt, agg, 0);

  for (int l = 0; l < NLAYERS; ++l) {
    const __bf16* WlT_l  = WlTb + (size_t)l * NET * HD * HD;
    const float*  bl_l   = bl   + (size_t)l * NET * HD;
    const __bf16* Wih_l  = Wihb + (size_t)l * TH3 * HD;
    const __bf16* Whh_l  = Whhb + (size_t)l * TH3 * HD;
    const float*  bih_l  = bih  + (size_t)l * TH3;
    const float*  bhh_l  = bhh  + (size_t)l * TH3;
    for (int s = 0; s < NSTEPS; ++s) {
      etype_gemm_kernel<<<GR_TILE, 32, 0, stream>>>(hb, WlT_l, bl_l, Whb);
      zero_kernel<<<GR_EL, 256, 0, stream>>>(aagg, NH);
      edge_scatter_kernel<<<GR_E32, 256, 0, stream>>>(src, dst, etype, Whb, aagg);
      cvt_f2b_kernel<<<GR_EL, 256, 0, stream>>>(aagg, ab, NH);
      gru_fused_kernel<<<GR_TILE, 32, 0, stream>>>(ab, Wih_l, Whh_l, bih_l, bhh_l, hfp, hb);
    }
    // intermediate[l+1] = mean_nodes(h)
    zero_kernel<<<(NG * HD + 255) / 256, 256, 0, stream>>>(sums, NG * HD);
    zero_kernel<<<1, 256, 0, stream>>>(cnt, NG);
    mean_accum_kernel<<<GR_N32, 256, 0, stream>>>(hfp, gid, sums, cnt);
    mean_div_kernel<<<(NG * HD + 255) / 256, 256, 0, stream>>>(sums, cnt, agg, (l + 1) * HD);
  }

  // head: out[0:64] = res, out[64:] = graph_agg
  head_kernel<<<NG, 128, 0, stream>>>(agg, W1, b1, W2, b2, out);
  copy_agg_kernel<<<(NG * TH3 + 255) / 256, 256, 0, stream>>>(agg, out + NG);
}
